// H2GCNBranch_58145267253791
// MI455X (gfx1250) — matrix operand order repaired and checked by reading, not compiled, not run on hardware
//
#include <hip/hip_runtime.h>

// H2GCN branch: h0 = x@W1 (bf16 WMMA, f32 acc), h1 = spmm(E1, h0), h2 = spmm(E2, h0)
// out = concat([h0, h1, h2], axis=1)  -> [N, 384] f32, row-major.
//
// Constants from the reference: IN_DIM=512, HID=128, out stride = 384.

typedef __bf16  v16bf __attribute__((ext_vector_type(16)));
typedef float   v8f   __attribute__((ext_vector_type(8)));

#define IN_DIM 512
#define HID    128
#define OSTR   384   // 3*HID output row stride

// ---------------------------------------------------------------------------
// Zero the whole output (h1/h2 are accumulated with atomics; harness poisons).
// ---------------------------------------------------------------------------
__global__ void h2gcn_zero_kernel(float* __restrict__ p, int n4) {
    int i = blockIdx.x * blockDim.x + threadIdx.x;
    int stride = gridDim.x * blockDim.x;
    float4 z = make_float4(0.f, 0.f, 0.f, 0.f);
    float4* p4 = (float4*)p;
    for (; i < n4; i += stride) p4[i] = z;
}

// ---------------------------------------------------------------------------
// GEMM: out[:, 0:128] = x[N,512] @ W1[512,128], bf16 WMMA 16x16x32, f32 acc.
// Block = 256 threads (8 waves). Wave w owns rows [blk*128 + w*16, +16) and
// the full 128-wide N dimension as 8 16x16 tiles.
// Per K-step (32): stage W1 tile transposed (bf16) in LDS, then 8 WMMAs.
// ---------------------------------------------------------------------------
__global__ void __launch_bounds__(256)
h2gcn_gemm_kernel(const float* __restrict__ x, const float* __restrict__ W1,
                  float* __restrict__ out, int Nrows) {
    // LDS: W1 K-tile transposed, [n][k] with n=0..127, k=0..31 (bf16) = 8 KB
    __shared__ __bf16 ldsW[HID * 32];

    const int tid     = threadIdx.x;
    const int lane    = tid & 31;
    const int wave    = tid >> 5;
    const int hs      = lane >> 4;   // half-select: 0 = lanes 0-15, 1 = lanes 16-31
    const int lrow    = lane & 15;

    const int rowBase = blockIdx.x * 128 + wave * 16;
    const bool valid  = (rowBase < Nrows);          // Nrows % 16 == 0 -> tile-exact
    const int  arow   = valid ? (rowBase + lrow) : lrow;  // clamp OOB waves' loads

    v8f acc[8] = {};   // 8 N-tiles x (16x16 f32 accumulator)

    const float* xrowbase = x + (size_t)arow * IN_DIM;

    for (int kb = 0; kb < IN_DIM / 32; ++kb) {
        // ---- stage W1[kb*32 .. kb*32+32, 0:128] into LDS, transposed, bf16 ----
        #pragma unroll
        for (int i = 0; i < (32 * HID) / 256; ++i) {
            int idx = tid + i * 256;
            int k = idx >> 7;        // 0..31
            int n = idx & 127;       // 0..127
            ldsW[n * 32 + k] = (__bf16)W1[(size_t)(kb * 32 + k) * HID + n];
        }
        __syncthreads();

        // ---- A fragment: contiguous k order (this code's choice, not the ISA table order) ----
        // lane<16: K = 0..15; lane>=16: K = 16..31
        v16bf a;
        {
            const float* xr = xrowbase + kb * 32 + hs * 16;
            const float4 p0 = *(const float4*)(xr + 0);
            const float4 p1 = *(const float4*)(xr + 4);
            const float4 p2 = *(const float4*)(xr + 8);
            const float4 p3 = *(const float4*)(xr + 12);
            a[0]  = (__bf16)p0.x; a[1]  = (__bf16)p0.y; a[2]  = (__bf16)p0.z; a[3]  = (__bf16)p0.w;
            a[4]  = (__bf16)p1.x; a[5]  = (__bf16)p1.y; a[6]  = (__bf16)p1.z; a[7]  = (__bf16)p1.w;
            a[8]  = (__bf16)p2.x; a[9]  = (__bf16)p2.y; a[10] = (__bf16)p2.z; a[11] = (__bf16)p2.w;
            a[12] = (__bf16)p3.x; a[13] = (__bf16)p3.y; a[14] = (__bf16)p3.z; a[15] = (__bf16)p3.w;
        }

        // ---- B fragments + WMMA over the 8 N-tiles ----
        // B in the same contiguous k order as A: lanes 0-15 hold K=0..15, lanes 16-31 hold K=16..31,
        // N = lane&15 -> 16 contiguous bf16 from the transposed LDS tile.
        #pragma unroll
        for (int nt = 0; nt < 8; ++nt) {
            const int n = nt * 16 + lrow;
            const v16bf b = *(const v16bf*)(&ldsW[n * 32 + hs * 16]);
            acc[nt] = __builtin_amdgcn_wmma_f32_16x16x32_bf16(
                /*neg_a=*/false, a, /*neg_b=*/false, b,
                /*c_mod=*/(short)0, acc[nt],
                /*reuse_a=*/false, /*reuse_b=*/false);
        }
        __syncthreads();   // before next K-step overwrites ldsW
    }

    // ---- store: C/D 16x16 f32 layout: VGPR r -> M = r + 8*hs, N = lane&15 ----
    if (valid) {
        #pragma unroll
        for (int nt = 0; nt < 8; ++nt) {
            const int col = nt * 16 + lrow;
            #pragma unroll
            for (int r = 0; r < 8; ++r) {
                const int m = rowBase + r + 8 * hs;
                out[(size_t)m * OSTR + col] = acc[nt][r];
            }
        }
    }
}

// ---------------------------------------------------------------------------
// SpMM: out[row[e], dstOff + :] += vals[e] * h0[col[e], :]
// h0 is read in-place from out columns [0,128). One wave32 per edge; each lane
// owns 4 contiguous hid-channels: 16B vector gather + 4 f32 atomics (L2 RMW).
// ---------------------------------------------------------------------------
__global__ void __launch_bounds__(256)
h2gcn_spmm_kernel(const int* __restrict__ row, const int* __restrict__ col,
                  const float* __restrict__ vals, float* __restrict__ out,
                  int E, int dstOff) {
    const int lane   = threadIdx.x & 31;
    const int wave   = (blockIdx.x * blockDim.x + threadIdx.x) >> 5;
    const int nwaves = (gridDim.x * blockDim.x) >> 5;

    for (int e = wave; e < E; e += nwaves) {
        const int   r = row[e];
        const int   c = col[e];
        const float v = vals[e];
        const float4 h = *(const float4*)(out + (size_t)c * OSTR + lane * 4);
        float* dst = out + (size_t)r * OSTR + dstOff + lane * 4;
        atomicAdd(dst + 0, v * h.x);
        atomicAdd(dst + 1, v * h.y);
        atomicAdd(dst + 2, v * h.z);
        atomicAdd(dst + 3, v * h.w);
    }
}

// ---------------------------------------------------------------------------
extern "C" void kernel_launch(void* const* d_in, const int* in_sizes, int n_in,
                              void* d_out, int out_size, void* d_ws, size_t ws_size,
                              hipStream_t stream) {
    const float* x     = (const float*)d_in[0];
    const float* W1    = (const float*)d_in[1];
    const int*   row1  = (const int*)  d_in[2];
    const int*   col1  = (const int*)  d_in[3];
    const float* vals1 = (const float*)d_in[4];
    const int*   row2  = (const int*)  d_in[5];
    const int*   col2  = (const int*)  d_in[6];
    const float* vals2 = (const float*)d_in[7];
    float* out = (float*)d_out;

    const int Nrows = in_sizes[0] / IN_DIM;   // 100000
    const int E1    = in_sizes[2];            // 3.2M
    const int E2    = in_sizes[5];            // 6.4M

    // 1) zero output (h1/h2 accumulated atomically; h0 overwritten by GEMM)
    h2gcn_zero_kernel<<<2048, 256, 0, stream>>>(out, out_size / 4);

    // 2) dense projection h0 = x @ W1 via bf16 WMMA
    h2gcn_gemm_kernel<<<(Nrows + 127) / 128, 256, 0, stream>>>(x, W1, out, Nrows);

    // 3) 1-hop and 2-hop aggregations (atomic scatter-add, L2-resident)
    h2gcn_spmm_kernel<<<4096, 256, 0, stream>>>(row1, col1, vals1, out, E1, HID);
    h2gcn_spmm_kernel<<<8192, 256, 0, stream>>>(row2, col2, vals2, out, E2, 2 * HID);
}
